// FutureEncoder_49950469653259
// MI455X (gfx1250) — matrix operand order repaired and checked by reading, not compiled, not run on hardware
//
#include <hip/hip_runtime.h>
#include <hip/hip_bf16.h>

// ---------------------------------------------------------------------------
// FutureEncoder fused forward for MI455X (gfx1250, wave32, WMMA).
// One wave = one scene (11 actors). All intermediates live in LDS; weights are
// repacked once (f16, transposed, K-padded) into d_ws by a prep kernel so that
// WMMA B-fragments are contiguous 16B loads (L2-resident, reused 8192x).
// ---------------------------------------------------------------------------

#define ACTORS 11
#define T_STEPS 10
#define WAVES_PER_BLOCK 4

typedef __attribute__((ext_vector_type(16))) _Float16 v16h;
typedef __attribute__((ext_vector_type(8)))  float    v8f;

// ---- input pointer indices (setup_inputs dict order, params depth-first) ----
enum : int {
  I_FM=0, I_FV=1, I_CTX=2, I_BS=3,
  I_W1=4, I_B1=5, I_W2=6, I_B2=7, I_W3=8, I_B3=9,
  I_EMB0W=10, I_EMB0B=11, I_EMB1W=12, I_EMB1B=13,
  I_L0E0W=14, I_L0E0B=15, I_L0E1W=16, I_L0E1B=17, I_L0NW=18, I_L0NB=19,
  I_L1E0W=20, I_L1E0B=21, I_L1E1W=22, I_L1E1B=23, I_L1NW=24, I_L1NB=25,
  I_IOW=26, I_IOB=27,
  I_H1N0W=28, I_H1N0B=29, I_H1N1W=30, I_H1N1B=31, I_H1OW=32, I_H1OB=33,
  I_H2N0W=34, I_H2N0B=35, I_H2N1W=36, I_H2N1B=37, I_H2OW=38, I_H2OB=39,
  I_MLPW=40, I_MLPB=41, I_QZW=42, I_QZB=43
};

struct PtrPack { const float* p[44]; };

// ---- half-element offsets of packed weights in d_ws (Wt layout [N][Kpad]) ----
enum : int {
  OFF_WCT   = 0,      // [16][64]   combined input_fc(+PE)+input_fc2
  OFF_W3T   = 1024,   // [16][32]
  OFF_EMB0T = 1536,   // [64][32]
  OFF_EMB1T = 3584,   // [16][64]
  OFF_E0T0  = 4608,   // [64][32]
  OFF_E1T0  = 6656,   // [16][64]
  OFF_NODET0= 7680,   // [16][32]
  OFF_E0T1  = 8192,
  OFF_E1T1  = 10240,
  OFF_NODET1= 11264,
  OFF_INTERT= 11776,  // [16][32]
  OFF_N0T0  = 12288,  // [64][32]
  OFF_N1T0  = 14336,  // [16][64]
  OFF_OUTT0 = 15360,  // [16][32]
  OFF_N0T1  = 15872,
  OFF_N1T1  = 17920,
  OFF_OUTT1 = 18944,
  OFF_MLPT  = 19456,  // [128][128]
  OFF_QZT   = 35840,  // [64][128]
  HALF_TOTAL= 44032
};
#define FLOAT_OFF_BYTES 88064  // HALF_TOTAL*2, 16B aligned

// ---- float offsets (biases) inside d_ws float region ----
enum : int {
  F_C0 = 0, F_B3E = 16, F_BEMB0 = 272, F_BEMB1 = 336,
  F_BE0_0 = 352, F_BE1_0 = 416, F_BNODE0 = 432,
  F_BE0_1 = 448, F_BE1_1 = 512, F_BNODE1 = 528,
  F_BINTER = 544,
  F_BN0_0 = 560, F_BN1_0 = 624, F_BOUT0 = 640,
  F_BN0_1 = 656, F_BN1_1 = 720, F_BOUT1 = 736,
  F_BMLP = 752, F_BQZ = 880
};

// ---------------------------------------------------------------------------
// wave-local LDS sync: LDS pipe is in-order per wave; we only need the DS
// counter drained + a compiler ordering fence before cross-lane reads.
__device__ inline void wave_sync() {
  asm volatile("s_wait_dscnt 0" ::: "memory");
  __builtin_amdgcn_wave_barrier();
}

// ---- WMMA fragment loaders (CDNA5 16-bit A / B lane layouts) ----
__device__ inline v16h load_frag_pair(const _Float16* r0, const _Float16* r1) {
  union { v16h v; uint4 q[2]; } u;
  u.q[0] = *(const uint4*)r0;
  u.q[1] = *(const uint4*)r1;
  return u.v;
}
// A 16x32 f16: lane m (0-15): K 0..7 (V0-3), 16..23 (V4-7); lane m+16: +8
__device__ inline v16h load_a_frag(const _Float16* Abuf, int ldk, int kc, int lane) {
  const int m  = lane & 15;
  const int kb = (lane & 16) ? 8 : 0;
  const _Float16* r = Abuf + m * ldk + kc + kb;
  return load_frag_pair(r, r + 16);
}
// B 32x16 f16: lane n (0-15): K 0..7, 16..23; lane n+16: +8
__device__ inline v16h load_b_frag(const _Float16* Wt, int ldk, int kc, int lane) {
  const int n  = lane & 15;
  const int kb = (lane & 16) ? 8 : 0;
  const _Float16* r = Wt + n * ldk + kc + kb;
  return load_frag_pair(r, r + 16);
}

// D = A[16,KC*32] x W[KC*32, NT*16], f32 accumulate
template<int NT, int KC>
__device__ inline void gemm(const _Float16* Alds, int ldk, const _Float16* Wt,
                            int lane, v8f* acc) {
  v16h af[KC];
#pragma unroll
  for (int kc = 0; kc < KC; ++kc) af[kc] = load_a_frag(Alds, ldk, kc * 32, lane);
#pragma unroll
  for (int nt = 0; nt < NT; ++nt) {
    v8f c = {};
#pragma unroll
    for (int kc = 0; kc < KC; ++kc) {
      v16h bf = load_b_frag(Wt + nt * 16 * ldk, ldk, kc * 32, lane);
      c = __builtin_amdgcn_wmma_f32_16x16x32_f16(false, af[kc], false, bf,
                                                 (short)0, c, false, false);
    }
    acc[nt] = c;
  }
}

// C/D layout: lane -> col n=lane&15, rows mb..mb+7 (mb = (lane&16)?8:0)
__device__ inline void store_tiles_f16(const v8f* acc, int nt, const float* bias,
                                       _Float16* dst, int ldd, int lane, bool doRelu) {
  const int n = lane & 15, mb = (lane & 16) ? 8 : 0;
  for (int t = 0; t < nt; ++t) {
    const float b = bias[t * 16 + n];
#pragma unroll
    for (int r = 0; r < 8; ++r) {
      float v = acc[t][r] + b;
      if (doRelu) v = fmaxf(v, 0.f);
      dst[(mb + r) * ldd + t * 16 + n] = (_Float16)v;
    }
  }
}

__device__ inline void build_A_pad(const _Float16* src, _Float16* A, int ldk, int lane) {
  for (int i = lane; i < 16 * ldk; i += 32) {
    const int m = i / ldk, k = i - m * ldk;
    A[i] = (k < 16) ? src[m * 16 + k] : (_Float16)0.f;
  }
}
__device__ inline void build_A_cat(const _Float16* s0, const _Float16* s1,
                                   _Float16* A, int lane) {
  for (int i = lane; i < 512; i += 32) {
    const int m = i >> 5, k = i & 31;
    A[i] = (k < 16) ? s0[m * 16 + k] : s1[m * 16 + (k - 16)];
  }
}

struct alignas(16) SceneLDS {
  _Float16 A[16 * 128];  // GEMM A staging (max K = 128)
  _Float16 E[128 * 16];  // all 121 edge messages
  _Float16 X[16 * 16];   // x
  _Float16 H[16 * 16];   // nmp hidden h
  _Float16 M[16 * 16];   // mean / hyper message
  _Float16 F[16 * 16];   // hyper node feature
  _Float16 I[16 * 16];   // inter output
  _Float16 G1[16 * 16];  // hyper1 output
  _Float16 G2[16 * 16];  // hyper2 output
  float    nrm[16];
};

// ---------------------------------------------------------------------------
// prep kernel: pack weights (f16 transposed, K padded to 32/64/128), fold
// input_fc+PE+input_fc2 into Wct/c0 and the one-hot into per-row bias b3eff.
// ---------------------------------------------------------------------------
__device__ inline void packWt(const float* W, int K, int N, int Kpad,
                              _Float16* dst, int tid, int nthr) {
  for (int idx = tid; idx < N * Kpad; idx += nthr) {
    const int n = idx / Kpad, k = idx - n * Kpad;
    dst[idx] = (_Float16)((k < K) ? W[k * N + n] : 0.f);
  }
}
__device__ inline void copyF(const float* src, int n, float* dst, int tid, int nthr) {
  for (int i = tid; i < n; i += nthr) dst[i] = src[i];
}

__global__ void __launch_bounds__(256) fe_prep(PtrPack pp, _Float16* H, float* Fb) {
  const int tid = threadIdx.x, nthr = blockDim.x;

  packWt(pp.p[I_W3],    16,  16,  32, H + OFF_W3T,    tid, nthr);
  packWt(pp.p[I_EMB0W], 16,  64,  32, H + OFF_EMB0T,  tid, nthr);
  packWt(pp.p[I_EMB1W], 64,  16,  64, H + OFF_EMB1T,  tid, nthr);
  packWt(pp.p[I_L0E0W], 32,  64,  32, H + OFF_E0T0,   tid, nthr);
  packWt(pp.p[I_L0E1W], 64,  16,  64, H + OFF_E1T0,   tid, nthr);
  packWt(pp.p[I_L0NW],  32,  16,  32, H + OFF_NODET0, tid, nthr);
  packWt(pp.p[I_L1E0W], 32,  64,  32, H + OFF_E0T1,   tid, nthr);
  packWt(pp.p[I_L1E1W], 64,  16,  64, H + OFF_E1T1,   tid, nthr);
  packWt(pp.p[I_L1NW],  32,  16,  32, H + OFF_NODET1, tid, nthr);
  packWt(pp.p[I_IOW],   16,  16,  32, H + OFF_INTERT, tid, nthr);
  packWt(pp.p[I_H1N0W], 16,  64,  32, H + OFF_N0T0,   tid, nthr);
  packWt(pp.p[I_H1N1W], 64,  16,  64, H + OFF_N1T0,   tid, nthr);
  packWt(pp.p[I_H1OW],  32,  16,  32, H + OFF_OUTT0,  tid, nthr);
  packWt(pp.p[I_H2N0W], 16,  64,  32, H + OFF_N0T1,   tid, nthr);
  packWt(pp.p[I_H2N1W], 64,  16,  64, H + OFF_N1T1,   tid, nthr);
  packWt(pp.p[I_H2OW],  32,  16,  32, H + OFF_OUTT1,  tid, nthr);
  packWt(pp.p[I_MLPW], 128, 128, 128, H + OFF_MLPT,   tid, nthr);
  packWt(pp.p[I_QZW],  128,  64, 128, H + OFF_QZT,    tid, nthr);

  copyF(pp.p[I_EMB0B], 64, Fb + F_BEMB0,  tid, nthr);
  copyF(pp.p[I_EMB1B], 16, Fb + F_BEMB1,  tid, nthr);
  copyF(pp.p[I_L0E0B], 64, Fb + F_BE0_0,  tid, nthr);
  copyF(pp.p[I_L0E1B], 16, Fb + F_BE1_0,  tid, nthr);
  copyF(pp.p[I_L0NB],  16, Fb + F_BNODE0, tid, nthr);
  copyF(pp.p[I_L1E0B], 64, Fb + F_BE0_1,  tid, nthr);
  copyF(pp.p[I_L1E1B], 16, Fb + F_BE1_1,  tid, nthr);
  copyF(pp.p[I_L1NB],  16, Fb + F_BNODE1, tid, nthr);
  copyF(pp.p[I_IOB],   16, Fb + F_BINTER, tid, nthr);
  copyF(pp.p[I_H1N0B], 64, Fb + F_BN0_0,  tid, nthr);
  copyF(pp.p[I_H1N1B], 16, Fb + F_BN1_0,  tid, nthr);
  copyF(pp.p[I_H1OB],  16, Fb + F_BOUT0,  tid, nthr);
  copyF(pp.p[I_H2N0B], 64, Fb + F_BN0_1,  tid, nthr);
  copyF(pp.p[I_H2N1B], 16, Fb + F_BN1_1,  tid, nthr);
  copyF(pp.p[I_H2OB],  16, Fb + F_BOUT1,  tid, nthr);
  copyF(pp.p[I_MLPB], 128, Fb + F_BMLP,   tid, nthr);
  copyF(pp.p[I_QZB],   64, Fb + F_BQZ,    tid, nthr);

  const float* w1 = pp.p[I_W1];  // [4,16]
  const float* b1 = pp.p[I_B1];
  const float* w2 = pp.p[I_W2];  // [160,16]
  const float* b2 = pp.p[I_B2];
  const float* w3 = pp.p[I_W3];  // [19,16]
  const float* b3 = pp.p[I_B3];

  // Wct[n][k], k = t*4+c : sum_j w1[c][j] * w2[t*16+j][n]
  for (int idx = tid; idx < 16 * 64; idx += nthr) {
    const int n = idx >> 6, k = idx & 63;
    float v = 0.f;
    if (k < 40) {
      const int t = k >> 2, c = k & 3;
      for (int j = 0; j < 16; ++j) v += w1[c * 16 + j] * w2[(t * 16 + j) * 16 + n];
    }
    H[OFF_WCT + idx] = (_Float16)v;
  }
  // c0[d] = b2[d] + sum_{t,j} (b1[j] + pe[t][j]) * w2[t*16+j][d]
  for (int d = tid; d < 16; d += nthr) {
    float acc = b2[d];
    for (int t = 0; t < T_STEPS; ++t)
      for (int j = 0; j < 16; ++j) {
        const int i2 = j >> 1;
        const float ang = (float)t * __expf(-(float)(2 * i2) * 0.57564627324851148f);
        const float pe = (j & 1) ? __cosf(ang) : __sinf(ang);
        acc += (b1[j] + pe) * w2[(t * 16 + j) * 16 + d];
      }
    Fb[F_C0 + d] = acc;
  }
  // b3eff[m][d] = b3[d] + w3[16 + onehot(m)][d]
  for (int idx = tid; idx < 256; idx += nthr) {
    const int m = idx >> 4, d = idx & 15;
    float v = b3[d];
    if (m < ACTORS) {
      const int cm = (m < 5) ? 0 : (m < 10 ? 1 : 2);
      v += w3[(16 + cm) * 16 + d];
    }
    Fb[F_B3E + idx] = v;
  }
}

// ---------------------------------------------------------------------------
// main kernel: one wave per scene
// ---------------------------------------------------------------------------
__global__ void __launch_bounds__(WAVES_PER_BLOCK * 32)
fe_main(PtrPack pp, const _Float16* W, const float* Fb, float* out, int nScenes) {
  __shared__ SceneLDS smem[WAVES_PER_BLOCK];
  const int lane = threadIdx.x & 31;
  const int wv   = threadIdx.x >> 5;
  const int s    = blockIdx.x * WAVES_PER_BLOCK + wv;
  if (s >= nScenes) return;            // wave-uniform guard (EXEC stays full)

  SceneLDS& L = smem[wv];
  _Float16* A = L.A;

  const float* fm  = pp.p[I_FM]  + (size_t)s * (ACTORS * T_STEPS * 2);
  const float* fv  = pp.p[I_FV]  + (size_t)s * (ACTORS * T_STEPS * 2);
  const float* ctx = pp.p[I_CTX] + (size_t)s * (ACTORS * 64);
  __builtin_prefetch(ctx, 0, 3);       // global_prefetch: ctx needed at the end

  // ---- stage 1: A_in[16][64] = traj (rows 11..15, cols 40..63 zero) ----
  for (int i = lane; i < 16 * 64; i += 32) A[i] = (_Float16)0.f;
  wave_sync();
  for (int e = lane; e < 220; e += 32) {
    const int n = e / 20, rem = e - n * 20, tt = rem >> 1, c = rem & 1;
    A[n * 64 + tt * 4 + c]     = (_Float16)fm[e];
    A[n * 64 + tt * 4 + 2 + c] = (_Float16)fv[e];
  }
  wave_sync();

  v8f acc[8];

  // ---- stage 2: x0 = A_in @ Wct + c0 ; x = x0 @ W3t + b3eff ----
  gemm<1, 2>(A, 64, W + OFF_WCT, lane, acc);
  for (int i = lane; i < 16 * 32; i += 32) A[i] = (_Float16)0.f;
  store_tiles_f16(acc, 1, Fb + F_C0, A, 32, lane, false);
  wave_sync();
  gemm<1, 1>(A, 32, W + OFF_W3T, lane, acc);
  {
    const int n = lane & 15, mb = (lane & 16) ? 8 : 0;
#pragma unroll
    for (int r = 0; r < 8; ++r) {
      const int m = mb + r;
      L.X[m * 16 + n] = (_Float16)(acc[0][r] + Fb[F_B3E + m * 16 + n]);
    }
  }
  wave_sync();

  // row norms of x (for feat_corr)
  if (lane < 16) {
    float s2 = 0.f;
    for (int d = 0; d < 16; ++d) { const float v = (float)L.X[lane * 16 + d]; s2 += v * v; }
    L.nrm[lane] = fmaxf(sqrtf(s2), 1e-12f);
  }
  wave_sync();

  // ---- nmp_ordinary: h = emb1(relu(emb0(x))) ----
  build_A_pad(L.X, A, 32, lane); wave_sync();
  gemm<4, 1>(A, 32, W + OFF_EMB0T, lane, acc);
  store_tiles_f16(acc, 4, Fb + F_BEMB0, A, 64, lane, true); wave_sync();
  gemm<1, 2>(A, 64, W + OFF_EMB1T, lane, acc);
  store_tiles_f16(acc, 1, Fb + F_BEMB1, L.H, 16, lane, false); wave_sync();

  for (int l = 0; l < 2; ++l) {
    const _Float16* e0t = W + (l ? OFF_E0T1 : OFF_E0T0);
    const _Float16* e1t = W + (l ? OFF_E1T1 : OFF_E1T0);
    const _Float16* ndt = W + (l ? OFF_NODET1 : OFF_NODET0);
    const float* be0 = Fb + (l ? F_BE0_1 : F_BE0_0);
    const float* be1 = Fb + (l ? F_BE1_1 : F_BE1_0);
    const float* bnd = Fb + (l ? F_BNODE1 : F_BNODE0);

    // 121 edges in 8 M-tiles of 16: e = i*11 + j -> [h_i | h_j] (K=32)
    for (int mt = 0; mt < 8; ++mt) {
      for (int i = lane; i < 512; i += 32) {
        const int r = i >> 5, k = i & 31, e = mt * 16 + r;
        _Float16 v = (_Float16)0.f;
        if (e < 121) {
          const int ii = e / 11, jj = e - ii * 11;
          v = (k < 16) ? L.H[ii * 16 + k] : L.H[jj * 16 + (k - 16)];
        }
        A[i] = v;
      }
      wave_sync();
      gemm<4, 1>(A, 32, e0t, lane, acc);
      store_tiles_f16(acc, 4, be0, A, 64, lane, true); wave_sync();
      gemm<1, 2>(A, 64, e1t, lane, acc);
      store_tiles_f16(acc, 1, be1, L.E + mt * 256, 16, lane, false);
      wave_sync();
    }
    // mean over j
    for (int p = lane; p < 176; p += 32) {
      const int i = p >> 4, d = p & 15;
      float sum = 0.f;
      for (int j = 0; j < ACTORS; ++j) sum += (float)L.E[(i * 11 + j) * 16 + d];
      L.M[p] = (_Float16)(sum * (1.f / 11.f));
    }
    for (int p = 176 + lane; p < 256; p += 32) L.M[p] = (_Float16)0.f;
    wave_sync();
    // h = node([h | m])
    build_A_cat(L.H, L.M, A, lane); wave_sync();
    gemm<1, 1>(A, 32, ndt, lane, acc);
    store_tiles_f16(acc, 1, bnd, L.H, 16, lane, false); wave_sync();
  }
  // inter = out(h)
  build_A_pad(L.H, A, 32, lane); wave_sync();
  gemm<1, 1>(A, 32, W + OFF_INTERT, lane, acc);
  store_tiles_f16(acc, 1, Fb + F_BINTER, L.I, 16, lane, false); wave_sync();

  // ---- nmp_hyper x2 (k = 5, 3) ----
  for (int hx = 0; hx < 2; ++hx) {
    const _Float16* n0t = W + (hx ? OFF_N0T1 : OFF_N0T0);
    const _Float16* n1t = W + (hx ? OFF_N1T1 : OFF_N1T0);
    const _Float16* ot  = W + (hx ? OFF_OUTT1 : OFF_OUTT0);
    const float* bn0 = Fb + (hx ? F_BN0_1 : F_BN0_0);
    const float* bn1 = Fb + (hx ? F_BN1_1 : F_BN1_0);
    const float* bo  = Fb + (hx ? F_BOUT1 : F_BOUT0);
    const int    K   = hx ? 3 : 5;
    _Float16*    G   = hx ? L.G2 : L.G1;

    build_A_pad(L.X, A, 32, lane); wave_sync();
    gemm<4, 1>(A, 32, n0t, lane, acc);
    store_tiles_f16(acc, 4, bn0, A, 64, lane, true); wave_sync();
    gemm<1, 2>(A, 64, n1t, lane, acc);
    store_tiles_f16(acc, 1, bn1, L.F, 16, lane, false); wave_sync();

    // per-actor top-k over corr row + softmax-weighted gather of f
    if (lane < 16) {
      if (lane < ACTORS) {
        const int i = lane;
        float v[ACTORS];
        for (int j = 0; j < ACTORS; ++j) {
          float d = 0.f;
          for (int t = 0; t < 16; ++t)
            d += (float)L.X[i * 16 + t] * (float)L.X[j * 16 + t];
          v[j] = d / (L.nrm[i] * L.nrm[j]);
        }
        int idxs[5]; float vals[5]; unsigned used = 0;
        for (int kk = 0; kk < K; ++kk) {
          int bi = 0; float bv = -3.4e38f;
          for (int j = 0; j < ACTORS; ++j)
            if (!((used >> j) & 1u) && v[j] > bv) { bv = v[j]; bi = j; }
          used |= (1u << bi); idxs[kk] = bi; vals[kk] = bv;
        }
        float w[5], se = 0.f;
        for (int kk = 0; kk < K; ++kk) { w[kk] = __expf(vals[kk] - vals[0]); se += w[kk]; }
        const float inv = 1.f / se;
        for (int d = 0; d < 16; ++d) {
          float m = 0.f;
          for (int kk = 0; kk < K; ++kk) m += w[kk] * (float)L.F[idxs[kk] * 16 + d];
          L.M[i * 16 + d] = (_Float16)(m * inv);
        }
      } else {
        for (int d = 0; d < 16; ++d) L.M[lane * 16 + d] = (_Float16)0.f;
      }
    }
    wave_sync();
    build_A_cat(L.X, L.M, A, lane); wave_sync();
    gemm<1, 1>(A, 32, ot, lane, acc);
    store_tiles_f16(acc, 1, bo, G, 16, lane, false); wave_sync();
  }

  // ---- final: [ctx | x | inter | h1 | h2] (K=128) -> relu MLP -> q_z ----
  for (int i = lane; i < 2048; i += 32) {
    const int m = i >> 7, k = i & 127;
    _Float16 v = (_Float16)0.f;
    if (m < ACTORS) {
      if (k < 64)        v = (_Float16)ctx[m * 64 + k];
      else if (k < 80)   v = L.X [m * 16 + (k - 64)];
      else if (k < 96)   v = L.I [m * 16 + (k - 80)];
      else if (k < 112)  v = L.G1[m * 16 + (k - 96)];
      else               v = L.G2[m * 16 + (k - 112)];
    }
    A[i] = v;
  }
  wave_sync();
  gemm<8, 4>(A, 128, W + OFF_MLPT, lane, acc);           // 32 WMMAs
  store_tiles_f16(acc, 8, Fb + F_BMLP, A, 128, lane, true);
  wave_sync();
  gemm<4, 4>(A, 128, W + OFF_QZT, lane, acc);            // 16 WMMAs
  {
    const int n = lane & 15, mb = (lane & 16) ? 8 : 0;
    for (int t = 0; t < 4; ++t) {
      const int ng = t * 16 + n;
      const float b = Fb[F_BQZ + ng];
#pragma unroll
      for (int r = 0; r < 8; ++r) {
        const int m = mb + r;
        if (m < ACTORS)
          out[((size_t)s * ACTORS + m) * 64 + ng] = acc[t][r] + b;
      }
    }
  }
}

// ---------------------------------------------------------------------------
extern "C" void kernel_launch(void* const* d_in, const int* in_sizes, int n_in,
                              void* d_out, int out_size, void* d_ws, size_t ws_size,
                              hipStream_t stream) {
  PtrPack pp{};
  const int np = (n_in < 44) ? n_in : 44;
  for (int i = 0; i < np; ++i) pp.p[i] = (const float*)d_in[i];

  const int B = in_sizes[0] / (ACTORS * T_STEPS * 2);

  _Float16* wsH = (_Float16*)d_ws;
  float*    wsF = (float*)((char*)d_ws + FLOAT_OFF_BYTES);

  fe_prep<<<1, 256, 0, stream>>>(pp, wsH, wsF);

  const int blocks = (B + WAVES_PER_BLOCK - 1) / WAVES_PER_BLOCK;
  fe_main<<<blocks, WAVES_PER_BLOCK * 32, 0, stream>>>(pp, wsH, wsF, (float*)d_out, B);
}
